// DualDeformModule_26079041421817
// MI455X (gfx1250) — matrix operand order repaired and checked by reading, not compiled, not run on hardware
//
#include <hip/hip_runtime.h>

typedef __attribute__((ext_vector_type(16))) _Float16 v16h;
typedef __attribute__((ext_vector_type(8)))  _Float16 v8h;
typedef __attribute__((ext_vector_type(8)))  float    v8f;

#define BATCH 4
#define CH    256
#define HW    4096       // 64*64
#define KTAP  9          // taps per channel (no padding)
#define KTOT  (CH*KTAP)  // 2304
#define NKCH  (KTOT/32)  // 72 k-chunks of 32

// ---------------------------------------------------------------------------
// Repack [O=256][C=256][9] f32 weights into WMMA B-fragment order (f16).
// Dest layout: [kt=0..71][nt=0..15][lane=0..31][half=0..15]
//   k_local = i + (lane>=16 ? 16 : 0) ; n = nt*16 + (lane&15)
//   k = kt*32 + k_local ; c = k/9 ; tap = k%9   (k < 2304, always valid)
// ---------------------------------------------------------------------------
__global__ void pack_weights(const float* __restrict__ w, _Float16* __restrict__ dst) {
  int tid = blockIdx.x * blockDim.x + threadIdx.x;       // 0 .. 589823
  int i  = tid & 15;
  int l  = (tid >> 4) & 31;
  int nt = (tid >> 9) & 15;
  int kt = tid >> 13;
  int kl = i + ((l >= 16) ? 16 : 0);
  int k  = kt * 32 + kl;
  int n  = (nt << 4) + (l & 15);
  int c  = k / 9;
  int tap = k - c * 9;
  dst[tid] = (_Float16)w[(size_t)(n * CH + c) * 9 + tap];
}

// ---------------------------------------------------------------------------
// Implicit-GEMM 3x3 conv, 256->256, pad 1, via v_wmma_f32_16x16x32_f16.
// Block = 128 threads (4 waves). M-tile = 32 positions (one row, x0..x0+31)
// => each B fragment feeds 2 WMMAs. Double-buffered LDS A staging (one
// barrier per chunk), branchless halo handling, vectorized ds_store_b128
// staging writes, global_prefetch on next chunk's B fragments.
// Output pre-GN, layout [b][m][c].
// ---------------------------------------------------------------------------
__global__ void conv3x3_gemm(const float* __restrict__ x,      // [B,C,64,64]
                             const _Float16* __restrict__ pw,  // packed B frags
                             const float* __restrict__ bias,   // [256]
                             float* __restrict__ pre) {        // [B,HW,256]
  __shared__ __align__(32) _Float16 ldsA[2][1024];   // 2 x (two 16x32 A tiles)
  const int lane = threadIdx.x & 31;
  const int wid  = threadIdx.x >> 5;
  const int b    = blockIdx.x >> 7;                  // 128 blocks per batch
  const int pos0 = (blockIdx.x << 5) & 4095;         // 32 positions per block
  const int yr   = pos0 >> 6;
  const int xc   = pos0 & 63;

  v8f acc[2][4];
  const v8f vz = {0.f,0.f,0.f,0.f,0.f,0.f,0.f,0.f};
  for (int mt = 0; mt < 2; ++mt)
    for (int j = 0; j < 4; ++j) acc[mt][j] = vz;

  // stage one k-chunk (32 k-values x 32 rows) into LDS in WMMA A layout;
  // each thread produces 8 contiguous halves -> one b128 LDS store.
  auto stage = [&](int buf, int kt) {
    const int e0 = threadIdx.x << 3;
    v8h hv;
#pragma unroll
    for (int q = 0; q < 8; ++q) {
      int e  = e0 + q;
      int mt = e >> 9;
      int r  = e & 511;
      int l  = r >> 4, i = r & 15;
      int kl = i + ((l >= 16) ? 16 : 0);
      int m  = (l & 15) + (mt << 4);
      int k  = kt * 32 + kl;
      int c  = k / 9;
      int tap = k - c * 9;
      int dy = tap / 3 - 1, dx = tap % 3 - 1;
      int yy = yr + dy;
      int xx = xc + m + dx;
      bool ok = ((unsigned)yy < 64u) & ((unsigned)xx < 64u);
      int idxp = ok ? ((yy << 6) + xx) : 0;
      float v = x[(((size_t)(b * CH + c)) << 12) + idxp];
      hv[q] = (_Float16)(ok ? v : 0.f);
    }
    *(v8h*)&ldsA[buf][e0] = hv;
  };

  stage(0, 0);
  for (int kt = 0; kt < NKCH; ++kt) {
    __syncthreads();                 // staged chunk kt visible in ldsA[kt&1]
    if (kt + 1 < NKCH) {
      // prefetch next chunk's B fragments for this wave (global_prefetch_b8)
      __builtin_prefetch(&pw[((size_t)((kt + 1) * 16 + (wid << 2)) << 9)], 0, 1);
      stage((kt + 1) & 1, kt + 1);   // overlap staging with WMMAs below
    }
    const int p = kt & 1;
    v16h a0 = *(const v16h*)&ldsA[p][lane << 4];
    v16h a1 = *(const v16h*)&ldsA[p][512 + (lane << 4)];
    for (int j = 0; j < 4; ++j) {
      int nt = (wid << 2) + j;
      const v16h bf = *(const v16h*)&pw[((size_t)((kt * 16 + nt) * 32 + lane)) << 4];
      acc[0][j] = __builtin_amdgcn_wmma_f32_16x16x32_f16(
          false, a0, false, bf, (short)0, acc[0][j], false, false);
      acc[1][j] = __builtin_amdgcn_wmma_f32_16x16x32_f16(
          false, a1, false, bf, (short)0, acc[1][j], false, false);
    }
  }

  const int mrow = (lane < 16) ? 0 : 8;
  for (int mt = 0; mt < 2; ++mt)
    for (int j = 0; j < 4; ++j) {
      int nt = (wid << 2) + j;
      int n  = (nt << 4) + (lane & 15);
      float bb = bias[n];
      for (int r = 0; r < 8; ++r) {
        int m = (mt << 4) + mrow + r;
        pre[(((size_t)(b * HW + pos0 + m)) << 8) + n] = acc[mt][j][r] + bb;
      }
    }
}

// ---------------------------------------------------------------------------
// Deformable 1x9 conv as implicit GEMM with on-the-fly bilinear sampling.
// off layout: [B][18][HW], channel 2k = y-offset, 2k+1 = x-offset.
// Same 32-row M-tile / double-buffer / prefetch / vectorized-staging
// structure as conv3x3_gemm; bilinear corners handled branchlessly.
// ---------------------------------------------------------------------------
__global__ void deform_gemm(const float* __restrict__ feat,   // [B,C,64,64]
                            const float* __restrict__ off,    // [B,18,HW]
                            const _Float16* __restrict__ pw,
                            float* __restrict__ pre) {        // [B,HW,256]
  __shared__ __align__(32) _Float16 ldsA[2][1024];
  __shared__ int   sy0[288], sx0[288];
  __shared__ float swy[288], swx[288];
  const int lane = threadIdx.x & 31;
  const int wid  = threadIdx.x >> 5;
  const int b    = blockIdx.x >> 7;
  const int pos0 = (blockIdx.x << 5) & 4095;
  const int yr   = pos0 >> 6;
  const int xc   = pos0 & 63;

  // precompute bilinear corner indices/weights: 32 positions x 9 points
  for (int e = threadIdx.x; e < 288; e += 128) {
    int m  = e / 9;
    int kk = e - m * 9;
    int pos = pos0 + m;
    float oy = off[(size_t)(b * 18 + 2 * kk)     * HW + pos];
    float ox = off[(size_t)(b * 18 + 2 * kk + 1) * HW + pos];
    float py = oy + (float)yr;
    float px = ox + (float)(xc + m) + (float)(kk - 4);
    float fy = floorf(py), fx = floorf(px);
    sy0[e] = (int)fy;
    sx0[e] = (int)fx;
    swy[e] = py - fy;
    swx[e] = px - fx;
  }
  __syncthreads();

  v8f acc[2][4];
  const v8f vz = {0.f,0.f,0.f,0.f,0.f,0.f,0.f,0.f};
  for (int mt = 0; mt < 2; ++mt)
    for (int j = 0; j < 4; ++j) acc[mt][j] = vz;

  auto stage = [&](int buf, int kt) {
    const int e0 = threadIdx.x << 3;
    v8h hv;
#pragma unroll
    for (int q = 0; q < 8; ++q) {
      int e  = e0 + q;
      int mt = e >> 9;
      int r  = e & 511;
      int l  = r >> 4, i = r & 15;
      int kl = i + ((l >= 16) ? 16 : 0);
      int m  = (l & 15) + (mt << 4);
      int k  = kt * 32 + kl;
      int c  = k / 9;
      int kk = k - c * 9;
      int t  = m * 9 + kk;
      int y0 = sy0[t], x0 = sx0[t];
      float wy = swy[t], wx = swx[t];
      const float* base = feat + (((size_t)(b * CH + c)) << 12);
      float s = 0.f;
#pragma unroll
      for (int cy = 0; cy < 2; ++cy)
#pragma unroll
        for (int cx = 0; cx < 2; ++cx) {
          int yy = y0 + cy, xx = x0 + cx;
          bool ok = ((unsigned)yy < 64u) & ((unsigned)xx < 64u);
          int idxp = ok ? ((yy << 6) + xx) : 0;
          float wgt = (cy ? wy : 1.f - wy) * (cx ? wx : 1.f - wx);
          float vv = base[idxp];
          s += ok ? (vv * wgt) : 0.f;
        }
      hv[q] = (_Float16)s;
    }
    *(v8h*)&ldsA[buf][e0] = hv;
  };

  stage(0, 0);
  for (int kt = 0; kt < NKCH; ++kt) {
    __syncthreads();
    if (kt + 1 < NKCH) {
      __builtin_prefetch(&pw[((size_t)((kt + 1) * 16 + (wid << 2)) << 9)], 0, 1);
      stage((kt + 1) & 1, kt + 1);
    }
    const int p = kt & 1;
    v16h a0 = *(const v16h*)&ldsA[p][lane << 4];
    v16h a1 = *(const v16h*)&ldsA[p][512 + (lane << 4)];
    for (int j = 0; j < 4; ++j) {
      int nt = (wid << 2) + j;
      const v16h bf = *(const v16h*)&pw[((size_t)((kt * 16 + nt) * 32 + lane)) << 4];
      acc[0][j] = __builtin_amdgcn_wmma_f32_16x16x32_f16(
          false, a0, false, bf, (short)0, acc[0][j], false, false);
      acc[1][j] = __builtin_amdgcn_wmma_f32_16x16x32_f16(
          false, a1, false, bf, (short)0, acc[1][j], false, false);
    }
  }

  const int mrow = (lane < 16) ? 0 : 8;
  for (int mt = 0; mt < 2; ++mt)
    for (int j = 0; j < 4; ++j) {
      int nt = (wid << 2) + j;
      int n  = (nt << 4) + (lane & 15);
      for (int r = 0; r < 8; ++r) {
        int m = (mt << 4) + mrow + r;
        pre[(((size_t)(b * HW + pos0 + m)) << 8) + n] = acc[mt][j][r];
      }
    }
}

// ---------------------------------------------------------------------------
// GroupNorm stats: one block per (b, g); reduce 8 channels x 4096 positions.
// pre layout [b][m][c]. stats[bg] = {mean, rstd}.
// ---------------------------------------------------------------------------
__global__ void gn_stats(const float* __restrict__ pre, float* __restrict__ stats) {
  __shared__ float ssum[256], ssq[256];
  int bg = blockIdx.x;            // 0..127
  int b = bg >> 5, g = bg & 31;
  float s = 0.f, q = 0.f;
  for (int e = threadIdx.x; e < 8 * HW; e += 256) {
    int m = e >> 3, cc = e & 7;
    float v = pre[(((size_t)(b * HW + m)) << 8) + (g << 3) + cc];
    s += v; q += v * v;
  }
  ssum[threadIdx.x] = s; ssq[threadIdx.x] = q;
  __syncthreads();
  for (int st = 128; st > 0; st >>= 1) {
    if (threadIdx.x < st) {
      ssum[threadIdx.x] += ssum[threadIdx.x + st];
      ssq[threadIdx.x]  += ssq[threadIdx.x + st];
    }
    __syncthreads();
  }
  if (threadIdx.x == 0) {
    float mean = ssum[0] * (1.f / 32768.f);
    float var  = ssq[0]  * (1.f / 32768.f) - mean * mean;
    stats[bg * 2]     = mean;
    stats[bg * 2 + 1] = rsqrtf(var + 1e-5f);
  }
}

// GN normalize + affine + ReLU. Input [b][m][c]; output same layout, or
// transposed to [b][c][m] (the reference NCHW output) when transpose_out!=0.
__global__ void gn_apply(const float* __restrict__ pre, const float* __restrict__ stats,
                         const float* __restrict__ gamma, const float* __restrict__ beta,
                         float* __restrict__ out, int transpose_out) {
  int idx = blockIdx.x * blockDim.x + threadIdx.x;   // < 4*4096*256
  int b = idx >> 20;
  int m = (idx >> 8) & 4095;
  int c = idx & 255;
  int bg = (b << 5) + (c >> 3);
  float mean = stats[bg * 2], rstd = stats[bg * 2 + 1];
  float v = (pre[idx] - mean) * rstd * gamma[c] + beta[c];
  v = fmaxf(v, 0.f);
  if (transpose_out) out[(((size_t)(b * CH + c)) << 12) + m] = v;
  else               out[idx] = v;
}

// ---------------------------------------------------------------------------
// Offset head: 3x3 conv 256->18 (direct VALU; ~1.4 GFLOP), emits pts into
// d_out ([B,18,64,64]) and off = pts - base_offset into scratch.
// ---------------------------------------------------------------------------
__global__ void offset_head(const float* __restrict__ t,     // [B,HW,256]
                            const float* __restrict__ w,     // [18,256,3,3]
                            const float* __restrict__ bias,  // [18]
                            float* __restrict__ pts,         // [B,18,HW]
                            float* __restrict__ offo) {      // [B,18,HW]
  int idx = blockIdx.x * blockDim.x + threadIdx.x;
  if (idx >= BATCH * 18 * HW) return;
  int pos = idx & 4095;
  int o   = (idx >> 12) % 18;
  int b   = idx / (18 * HW);
  int y = pos >> 6, x = pos & 63;
  float acc = bias[o];
  for (int tap = 0; tap < 9; ++tap) {
    int dy = tap / 3 - 1, dx = tap % 3 - 1;
    int yy = y + dy, xx = x + dx;
    if (yy < 0 || yy >= 64 || xx < 0 || xx >= 64) continue;
    const float* trow = t + (((size_t)(b * HW + (yy << 6) + xx)) << 8);
    const float* wrow = w + (size_t)(o * CH) * 9 + tap;
    float s = 0.f;
    for (int c = 0; c < CH; ++c) s += trow[c] * wrow[c * 9];
    acc += s;
  }
  pts[idx] = acc;
  float base = (o & 1) ? (float)((o >> 1) - 4) : 0.f;
  offo[idx] = acc - base;
}

// ---------------------------------------------------------------------------
extern "C" void kernel_launch(void* const* d_in, const int* in_sizes, int n_in,
                              void* d_out, int out_size, void* d_ws, size_t ws_size,
                              hipStream_t stream) {
  (void)in_sizes; (void)n_in; (void)out_size; (void)ws_size;
  const float* cls_feat = (const float*)d_in[0];
  const float* reg_feat = (const float*)d_in[1];
  const float* offc_w   = (const float*)d_in[2];
  const float* offc_b   = (const float*)d_in[3];
  const float* offc_g   = (const float*)d_in[4];
  const float* offc_bt  = (const float*)d_in[5];
  const float* offo_w   = (const float*)d_in[6];
  const float* offo_b   = (const float*)d_in[7];
  const float* clsdc_w  = (const float*)d_in[8];
  const float* cls_g    = (const float*)d_in[9];
  const float* cls_bt   = (const float*)d_in[10];
  const float* regdc_w  = (const float*)d_in[11];
  const float* reg_g    = (const float*)d_in[12];
  const float* reg_bt   = (const float*)d_in[13];
  float* out = (float*)d_out;

  char* ws = (char*)d_ws;
  float*    pre   = (float*)   (ws + 0);          // 16 MB   [B,HW,C] pre-GN
  float*    t     = (float*)   (ws + 16777216);   // 16 MB   [B,HW,C] after GN+ReLU
  float*    off   = (float*)   (ws + 33554432);   // 1.1 MB  [B,18,HW]
  _Float16* pw1   = (_Float16*)(ws + 34734080);   // 1.125MB packed conv1 weights
  _Float16* pw2   = (_Float16*)(ws + 35913728);   // 1.125MB packed cls dcn weights
  _Float16* pw3   = (_Float16*)(ws + 37093376);   // 1.125MB packed reg dcn weights
  float*    stats = (float*)   (ws + 38273024);   // 1 KB    {mean,rstd} x 128

  pack_weights<<<2304, 256, 0, stream>>>(offc_w,  pw1);
  pack_weights<<<2304, 256, 0, stream>>>(clsdc_w, pw2);
  pack_weights<<<2304, 256, 0, stream>>>(regdc_w, pw3);

  // branch 1: conv3x3 -> GN -> ReLU
  conv3x3_gemm<<<512, 128, 0, stream>>>(reg_feat, pw1, offc_b, pre);
  gn_stats<<<128, 256, 0, stream>>>(pre, stats);
  gn_apply<<<16384, 256, 0, stream>>>(pre, stats, offc_g, offc_bt, t, 0);

  // offset head -> pts (output 0) and off scratch
  offset_head<<<1152, 256, 0, stream>>>(t, offo_w, offo_b, out, off);

  // cls deform conv -> GN -> ReLU (output 1, NCHW)
  deform_gemm<<<512, 128, 0, stream>>>(cls_feat, off, pw2, pre);
  gn_stats<<<128, 256, 0, stream>>>(pre, stats);
  gn_apply<<<16384, 256, 0, stream>>>(pre, stats, cls_g, cls_bt, out + 294912, 1);

  // reg deform conv -> GN -> ReLU (output 2, NCHW)
  deform_gemm<<<512, 128, 0, stream>>>(reg_feat, off, pw3, pre);
  gn_stats<<<128, 256, 0, stream>>>(pre, stats);
  gn_apply<<<16384, 256, 0, stream>>>(pre, stats, reg_g, reg_bt, out + 294912 + 4194304, 1);
}
